// TreeEncoderRNN_712964571853
// MI455X (gfx1250) — matrix operand order repaired and checked by reading, not compiled, not run on hardware
//
#include <hip/hip_runtime.h>
#include <hip/hip_bf16.h>
#include <stdint.h>

typedef __attribute__((ext_vector_type(16))) __bf16 v16bf;
typedef __attribute__((ext_vector_type(8)))  __bf16 v8bf;
typedef __attribute__((ext_vector_type(8)))  float  v8f;

#define HDIM 256
#define SEQ  2048
#define NLEV 11
#define BM   64

__device__ __forceinline__ __bf16 f2bf(float f) {
  union { float f; uint32_t u; } x; x.f = f;
  uint32_t r = (x.u + 0x7FFFu + ((x.u >> 16) & 1u)) >> 16;   // round-to-nearest-even
  union { unsigned short s; __bf16 b; } y; y.s = (unsigned short)r;
  return y.b;
}
__device__ __forceinline__ float bf2f(__bf16 b) {
  union { unsigned short s; __bf16 b; } y; y.b = b;
  union { uint32_t u; float f; } x; x.u = ((uint32_t)y.s) << 16;
  return x.f;
}

// Pack weights: Wcat[set][n=head*256+j][k=0..511] (bf16), k<256 -> *_l, k>=256 -> *_r.
// set 0 = level-1 (w_*), set 1 = levels>=2 (u_*). bias[head*256+j] = sum of all 4 biases.
__global__ __launch_bounds__(256) void prep_weights(
    const float* __restrict__ wzl, const float* __restrict__ wzr,
    const float* __restrict__ whl, const float* __restrict__ whr,
    const float* __restrict__ uzl, const float* __restrict__ uzr,
    const float* __restrict__ uhl, const float* __restrict__ uhr,
    const float* __restrict__ bwzl, const float* __restrict__ bwzr,
    const float* __restrict__ buzl, const float* __restrict__ buzr,
    const float* __restrict__ bwhl, const float* __restrict__ bwhr,
    const float* __restrict__ buhl, const float* __restrict__ buhr,
    __bf16* __restrict__ Wcat, float* __restrict__ bias)
{
  int idx = blockIdx.x * 256 + threadIdx.x;          // 0 .. 2*512*512-1
  int set  = idx >> 18;
  int n    = (idx >> 9) & 511;
  int k    = idx & 511;
  int head = n >> 8;
  int j    = n & 255;
  int kk   = k & 255;
  const float* src;
  if (set == 0) src = head ? (k < 256 ? whl : whr) : (k < 256 ? wzl : wzr);
  else          src = head ? (k < 256 ? uhl : uhr) : (k < 256 ? uzl : uzr);
  Wcat[idx] = f2bf(src[j * 256 + kk]);
  if (idx < 512) {
    int h2 = idx >> 8, jj = idx & 255;
    bias[idx] = h2 ? (bwhl[jj] + bwhr[jj] + buhl[jj] + buhr[jj])
                   : (bwzl[jj] + bwzr[jj] + buzl[jj] + buzr[jj]);
  }
}

// One tree level: rows m in [0,M); (b,p) = (m>>logP, m & (P-1)).
// A[m] = bf16([x_l ‖ x_r]) (K=512). z = sig(A@Wz^T+bz), ht = sig(A@Wh^T+bh),
// h = z*(x_l+x_r) + (1-z)*ht   (gated term 0 at level 1).
// Block = 8 waves; wave = 2 M-tiles (32 rows) x 64 cols of BOTH heads (8 N-tiles).
__global__ __launch_bounds__(256) void tree_gemm(
    int level, int logP,
    const int*    __restrict__ tokens,
    const float*  __restrict__ emb,
    const __bf16* __restrict__ xin,
    const __bf16* __restrict__ W,     // [512][512] bf16, row n = head*256+j
    const float*  __restrict__ bias,  // [512]
    __bf16* __restrict__ hout,
    float*  __restrict__ fout)
{
  __shared__ alignas(16) __bf16 aT[BM][512 + 8];
  __shared__ uintptr_t srcLa[BM], srcRa[BM];

  const int tid = threadIdx.x;
  const int m0  = blockIdx.x * BM;

  if (tid < BM) {
    int m = m0 + tid;
    int b = m >> logP;
    int p = m & ((1 << logP) - 1);
    if (level == 1) {
      int t0 = tokens[b * SEQ + 2 * p];
      int t1 = tokens[b * SEQ + 2 * p + 1];
      srcLa[tid] = (uintptr_t)(emb + (size_t)t0 * HDIM);
      srcRa[tid] = (uintptr_t)(emb + (size_t)t1 * HDIM);
    } else {
      size_t r0 = ((size_t)b << (logP + 1)) + 2 * (size_t)p;
      srcLa[tid] = (uintptr_t)(xin + r0 * HDIM);
      srcRa[tid] = (uintptr_t)(xin + (r0 + 1) * HDIM);
    }
  }
  __syncthreads();

  // Stage A tile: 64 rows x 512 bf16, 8 elements per thread per iteration.
  #pragma unroll
  for (int it = 0; it < 16; ++it) {
    int e  = (tid + it * 256) << 3;
    int m  = e >> 9;
    int k  = e & 511;
    int kk = k & 255;
    uintptr_t src = (k < 256) ? srcLa[m] : srcRa[m];
    __bf16* dst = &aT[m][k];
    if (level == 1) {
      const float* s = (const float*)src + kk;
      #pragma unroll
      for (int i = 0; i < 8; ++i) dst[i] = f2bf(s[i]);
    } else {
      *(v8bf*)dst = *(const v8bf*)((const __bf16*)src + kk);
    }
  }
  __syncthreads();

  const int lane  = tid & 31;
  const int wv    = tid >> 5;     // 8 waves
  const int mHalf = wv >> 2;      // rows 32*mHalf .. +31 (two 16-row tiles)
  const int nQuad = wv & 3;       // 64-col slice of each head
  const int r     = lane & 15;
  const int hi    = lane >> 4;
  const int row0  = mHalf * 32 + r;
  const int row1  = row0 + 16;

  v8f acc[2][8] = {};             // [m-tile][nt]; nt 0..3: z head, nt 4..7: h~ head

  for (int k0 = 0; k0 < 512; k0 += 32) {
    // A fragments (K contiguous per lane, K = k0 + hi*16 + i): two contiguous 16B LDS reads each.
    union Af { v16bf v; struct { v8bf lo, up; } s; } a0, a1;
    a0.s.lo = *(const v8bf*)&aT[row0][k0 + hi * 16];
    a0.s.up = *(const v8bf*)&aT[row0][k0 + hi * 16 + 8];
    a1.s.lo = *(const v8bf*)&aT[row1][k0 + hi * 16];
    a1.s.up = *(const v8bf*)&aT[row1][k0 + hi * 16 + 8];
    // B fragments (K contiguous per lane): N = nBase + (lane&15), K = k0 + hi*16 + i.
    const __bf16* wp0 = W + (size_t)(nQuad * 64 + r) * 512 + k0 + hi * 16;
    const __bf16* wp1 = wp0 + (size_t)256 * 512;      // h~ head
    #pragma unroll
    for (int nt = 0; nt < 4; ++nt) {
      v16bf bz = *(const v16bf*)(wp0 + (size_t)nt * 16 * 512);
      acc[0][nt] = __builtin_amdgcn_wmma_f32_16x16x32_bf16(
          false, a0.v, false, bz, (short)0, acc[0][nt], false, false);
      acc[1][nt] = __builtin_amdgcn_wmma_f32_16x16x32_bf16(
          false, a1.v, false, bz, (short)0, acc[1][nt], false, false);
      v16bf bh = *(const v16bf*)(wp1 + (size_t)nt * 16 * 512);
      acc[0][nt + 4] = __builtin_amdgcn_wmma_f32_16x16x32_bf16(
          false, a0.v, false, bh, (short)0, acc[0][nt + 4], false, false);
      acc[1][nt + 4] = __builtin_amdgcn_wmma_f32_16x16x32_bf16(
          false, a1.v, false, bh, (short)0, acc[1][nt + 4], false, false);
    }
  }

  // Epilogue entirely in registers: lane holds out[m][j] and out[m][256+j] aligned.
  const bool last = (level == NLEV);
  #pragma unroll
  for (int mt = 0; mt < 2; ++mt) {
    #pragma unroll
    for (int nt = 0; nt < 4; ++nt) {
      int j  = nQuad * 64 + nt * 16 + r;      // 0..255
      float bz = bias[j];
      float bh = bias[256 + j];
      #pragma unroll
      for (int i = 0; i < 8; ++i) {
        int m = mHalf * 32 + mt * 16 + hi * 8 + i;   // C/D layout: M = hi*8+i, N = lane&15
        float z  = 1.0f / (1.0f + __expf(-(acc[mt][nt][i]     + bz)));
        float ht = 1.0f / (1.0f + __expf(-(acc[mt][nt + 4][i] + bh)));
        float s  = 0.0f;
        if (level > 1) s = bf2f(aT[m][j]) + bf2f(aT[m][256 + j]);   // h_l + h_r
        float h = z * s + (1.0f - z) * ht;
        size_t o = (size_t)(m0 + m) * HDIM + j;
        if (last) fout[o] = h;
        else      hout[o] = f2bf(h);
      }
    }
  }
}

extern "C" void kernel_launch(void* const* d_in, const int* in_sizes, int n_in,
                              void* d_out, int out_size, void* d_ws, size_t ws_size,
                              hipStream_t stream) {
  (void)in_sizes; (void)n_in; (void)out_size; (void)ws_size;
  const int*   tokens = (const int*)d_in[0];
  const float* emb    = (const float*)d_in[1];

  char* ws = (char*)d_ws;
  __bf16* Wcat = (__bf16*)ws;                                   // 2 sets * 512*512 bf16 = 1 MiB
  float*  bias = (float*)(ws + (size_t)(1u << 20));             // 512 f32
  __bf16* h0   = (__bf16*)(ws + (size_t)(2u << 20));            // 128 MiB (odd levels out)
  __bf16* h1   = (__bf16*)(ws + (size_t)(2u << 20) + (size_t)134217728u); // 64 MiB

  prep_weights<<<2048, 256, 0, stream>>>(
      (const float*)d_in[2],  (const float*)d_in[4],   // w_z_l, w_z_r
      (const float*)d_in[10], (const float*)d_in[12],  // w_h_l, w_h_r
      (const float*)d_in[6],  (const float*)d_in[8],   // u_z_l, u_z_r
      (const float*)d_in[14], (const float*)d_in[16],  // u_h_l, u_h_r
      (const float*)d_in[3],  (const float*)d_in[5],   // b_w_z_l, b_w_z_r
      (const float*)d_in[7],  (const float*)d_in[9],   // b_u_z_l, b_u_z_r
      (const float*)d_in[11], (const float*)d_in[13],  // b_w_h_l, b_w_h_r
      (const float*)d_in[15], (const float*)d_in[17],  // b_u_h_l, b_u_h_r
      Wcat, bias);

  int M = (256 * 2048) / 2;                 // level-1 output rows
  const __bf16* xin = nullptr;
  for (int l = 1; l <= NLEV; ++l) {
    int logP = NLEV - l;                    // pairs-per-batch = 2^logP
    __bf16* hout = (l & 1) ? h0 : h1;
    const __bf16* Wsel = Wcat + (size_t)(l == 1 ? 0 : 512 * 512);
    tree_gemm<<<dim3(M / BM), dim3(256), 0, stream>>>(
        l, logP, tokens, emb, xin, Wsel, bias, hout, (float*)d_out);
    xin = hout;
    M >>= 1;
  }
}
